// HGSA_v7_86947317940638
// MI455X (gfx1250) — matrix operand order repaired and checked by reading, not compiled, not run on hardware
//
#include <hip/hip_runtime.h>
#include <hip/hip_bf16.h>
#include <math.h>

// ---------------------------------------------------------------------------
// Types for CDNA5 WMMA (wave32, 16x16x32 bf16 -> f32)
// ---------------------------------------------------------------------------
typedef __attribute__((ext_vector_type(16))) __bf16 v16bf;
typedef __attribute__((ext_vector_type(8)))  float  v8f;

#define ACT_NONE 0
#define ACT_SILU 1
#define ACT_SIGM 2

__device__ __forceinline__ float actf(float v, int act) {
  if (act == ACT_SILU) return v / (1.f + expf(-v));
  if (act == ACT_SIGM) return 1.f / (1.f + expf(-v));
  return v;
}

// Native f32 -> bf16 convert (backend picks v_cvt_*_bf16 on gfx1250).
__device__ __forceinline__ __bf16 to_bf16(float f) { return (__bf16)f; }

// ---------------------------------------------------------------------------
// WMMA pointwise (1x1) conv:  Y[b,co,n] = act( sum_ci W[co,ci]*X[b,ci,n] + B[co] (+R) )
// One wave computes a 16(M=Cout) x 64(N=pixels) tile: 4 accumulators share one
// A (weight) fragment per K-step; K unrolled at compile time (Cin <= 64 here).
// Fragments are built directly from global memory (each element is consumed
// exactly once, so LDS staging is pure overhead; data is L2-resident).
// Tail handling: clamp-index + select (v_cndmask) — never a guarded load, so
// no EXEC-mask branching. KFULL (Cin==32*KSTEPS) / MFULL (Cout%16==0) /
// HASR (residual present) specializations drop all runtime guards for the
// hot 32-channel GEMMs.
// k order used here (contiguous, same in A and B): A lane<16 holds K 0-15 of row M=lane%16; lane>=16 holds
// K 16-31. B lane half selects K 0-15 / 16-31, N = lane%16.
// D: VGPR r -> M = r + 8*(lane>=16), N = lane%16.
// ---------------------------------------------------------------------------
template <int KSTEPS, bool KFULL, bool MFULL, bool HASR>
__global__ void __launch_bounds__(128)
k_pw_wmma(const float* __restrict__ X, const float* __restrict__ Wt,
          const float* __restrict__ Bias, const float* __restrict__ Rin,
          float* __restrict__ Y, int Cin, int Cout, int lgHW, int Bn, int act)
{
  const int wave = threadIdx.x >> 5;
  const int lane = threadIdx.x & 31;
  const int col  = lane & 15;
  const int half = lane >> 4;
  const long HW   = 1L << lgHW;
  const long Ntot = (long)Bn << lgHW;            // multiple of 64 here
  const long n0   = ((long)blockIdx.x * 4 + wave) * 64;
  if (n0 >= Ntot) return;                        // no barriers in this kernel
  const int  m0  = blockIdx.y * 16;
  const long b   = n0 >> lgHW;                   // tile never straddles batches
  const long rem = n0 & (HW - 1);
  const float* xb = X + (((size_t)b * Cin) << lgHW) + rem;

  if (KSTEPS > 1)                                // hint next K block into cache
    __builtin_prefetch(xb + ((size_t)(32 + lane) << lgHW), 0, 1); // global_prefetch_b8

  v8f acc[4] = {};
#pragma unroll
  for (int ks = 0; ks < KSTEPS; ++ks) {
    const int kk = ks << 5;
    // ---- A fragment (weights 16x32) ----
    v16bf afrag;
    const int m = m0 + col;
#pragma unroll
    for (int e = 0; e < 16; ++e) {
      int klocal = e + (half ? 16 : 0);
      int kidx = kk + klocal;
      float wv;
      if (KFULL && MFULL) {
        wv = Wt[(size_t)m * Cin + kidx];
      } else {
        int mc = MFULL ? m : min(m, Cout - 1);
        int kc = KFULL ? kidx : min(kidx, Cin - 1);
        wv = Wt[(size_t)mc * Cin + kc];           // always in-bounds
        bool pad = (!MFULL && m >= Cout) || (!KFULL && kidx >= Cin);
        wv = pad ? 0.f : wv;                      // v_cndmask, no branch
      }
      afrag[e] = to_bf16(wv);
    }
    // ---- 4 B fragments: one row pointer per k, 4 loads at immediate offsets ----
    const int kbase = kk + half * 16;
    v16bf bf0, bf1, bf2, bf3;
#pragma unroll
    for (int e = 0; e < 16; ++e) {
      int kidx = kbase + e;
      int kc = KFULL ? kidx : min(kidx, Cin - 1);
      const float* rp = xb + ((size_t)kc << lgHW) + col;
      float x0v = rp[0], x1v = rp[16], x2v = rp[32], x3v = rp[48];
      if (!KFULL) {
        bool pad = kidx >= Cin;
        x0v = pad ? 0.f : x0v; x1v = pad ? 0.f : x1v;
        x2v = pad ? 0.f : x2v; x3v = pad ? 0.f : x3v;
      }
      bf0[e] = to_bf16(x0v); bf1[e] = to_bf16(x1v);
      bf2[e] = to_bf16(x2v); bf3[e] = to_bf16(x3v);
    }
    acc[0] = __builtin_amdgcn_wmma_f32_16x16x32_bf16(false, afrag, false, bf0,
                                                     (short)0, acc[0], false, false);
    acc[1] = __builtin_amdgcn_wmma_f32_16x16x32_bf16(false, afrag, false, bf1,
                                                     (short)0, acc[1], false, false);
    acc[2] = __builtin_amdgcn_wmma_f32_16x16x32_bf16(false, afrag, false, bf2,
                                                     (short)0, acc[2], false, false);
    acc[3] = __builtin_amdgcn_wmma_f32_16x16x32_bf16(false, afrag, false, bf3,
                                                     (short)0, acc[3], false, false);
  }
  float* yb = Y + (((size_t)b * Cout) << lgHW) + rem;
  const float* rb = HASR ? Rin + (((size_t)b * Cout) << lgHW) + rem : nullptr;
#pragma unroll
  for (int s = 0; s < 4; ++s) {
#pragma unroll
    for (int r = 0; r < 8; ++r) {
      int m = m0 + r + (half << 3);
      if (MFULL || m < Cout) {
        size_t o = ((size_t)m << lgHW) + s * 16 + col;
        float v = acc[s][r] + Bias[m];
        if (HASR) v += rb[o];
        yb[o] = actf(v, act);
      }
    }
  }
}

// ---------------------------------------------------------------------------
// Generic direct conv (stride/pad/dilation/groups) — dw & kxk convs
// ---------------------------------------------------------------------------
__global__ void k_conv2d(const float* __restrict__ x, const float* __restrict__ w,
                         const float* __restrict__ bias, float* __restrict__ y,
                         int Bn, int Ci, int Hin, int Win, int Co, int K,
                         int stride, int pad, int dil, int groups, int act,
                         int Hout, int Wout)
{
  int idx = blockIdx.x * blockDim.x + threadIdx.x;
  int total = Bn * Co * Hout * Wout;
  if (idx >= total) return;
  int wo = idx % Wout; int t = idx / Wout;
  int ho = t % Hout; t /= Hout;
  int co = t % Co;  int b = t / Co;
  int cpg_i = Ci / groups, cpg_o = Co / groups;
  int g = co / cpg_o;
  float acc = bias[co];
  for (int ci = 0; ci < cpg_i; ++ci) {
    int cin = g * cpg_i + ci;
    for (int kh = 0; kh < K; ++kh) {
      int hi = ho * stride - pad + kh * dil;
      if (hi < 0 || hi >= Hin) continue;
      for (int kw = 0; kw < K; ++kw) {
        int wi = wo * stride - pad + kw * dil;
        if (wi < 0 || wi >= Win) continue;
        acc += x[((size_t)(b * Ci + cin) * Hin + hi) * Win + wi] *
               w[(((size_t)co * cpg_i + ci) * K + kh) * K + kw];
      }
    }
  }
  y[idx] = actf(acc, act);
}

// LayerNorm over channel dim, per pixel (HW = 1<<lgHW)
__global__ void k_layernorm(const float* __restrict__ x, const float* __restrict__ gg,
                            const float* __restrict__ bb, float* __restrict__ y,
                            int Bn, int C, int lgHW)
{
  int idx = blockIdx.x * blockDim.x + threadIdx.x;
  int HW = 1 << lgHW;
  if (idx >= (Bn << lgHW)) return;
  int b = idx >> lgHW, rem = idx & (HW - 1);
  const float* xp = x + (((size_t)b * C) << lgHW) + rem;
  float mu = 0.f;
  for (int c = 0; c < C; ++c) mu += xp[(size_t)c << lgHW];
  mu /= C;
  float var = 0.f;
  for (int c = 0; c < C; ++c) { float d = xp[(size_t)c << lgHW] - mu; var += d * d; }
  var /= C;
  float inv = rsqrtf(var + 1e-5f);
  float* yp = y + (((size_t)b * C) << lgHW) + rem;
  for (int c = 0; c < C; ++c)
    yp[(size_t)c << lgHW] = (xp[(size_t)c << lgHW] - mu) * inv * gg[c] + bb[c];
}

// mean over spatial dims: out[row] = mean(x[row, 0..HW))
__global__ void k_spatial_mean(const float* __restrict__ x, float* __restrict__ out, int HW)
{
  __shared__ float red[256];
  const float* xp = x + (size_t)blockIdx.x * HW;
  float s = 0.f;
  for (int i = threadIdx.x; i < HW; i += 256) s += xp[i];
  red[threadIdx.x] = s; __syncthreads();
  for (int st = 128; st > 0; st >>= 1) {
    if ((int)threadIdx.x < st) red[threadIdx.x] += red[threadIdx.x + st];
    __syncthreads();
  }
  if (threadIdx.x == 0) out[blockIdx.x] = red[0] / (float)HW;
}

// 1x1 conv on [B, C] vectors (global-pooled features)
__global__ void k_vec_conv(const float* __restrict__ in, const float* __restrict__ w,
                           const float* __restrict__ bias, float* __restrict__ out,
                           int Bn, int Ci, int Co, int act)
{
  int idx = blockIdx.x * blockDim.x + threadIdx.x;
  if (idx >= Bn * Co) return;
  int co = idx % Co, b = idx / Co;
  float a = bias[co];
  for (int ci = 0; ci < Ci; ++ci) a += w[co * Ci + ci] * in[b * Ci + ci];
  out[idx] = actf(a, act);
}

// in-place L2 normalize each row of length n
__global__ void k_l2norm_rows(float* __restrict__ x, int n)
{
  __shared__ float red[256];
  __shared__ float inv;
  float* xp = x + (size_t)blockIdx.x * n;
  float s = 0.f;
  for (int i = threadIdx.x; i < n; i += 256) { float v = xp[i]; s += v * v; }
  red[threadIdx.x] = s; __syncthreads();
  for (int st = 128; st > 0; st >>= 1) {
    if ((int)threadIdx.x < st) red[threadIdx.x] += red[threadIdx.x + st];
    __syncthreads();
  }
  if (threadIdx.x == 0) inv = 1.f / fmaxf(sqrtf(red[0]), 1e-12f);
  __syncthreads();
  for (int i = threadIdx.x; i < n; i += 256) xp[i] *= inv;
}

// out[((b*heads+h)*r1+i)*r2+j] = temp[h] * sum_n A[b, h*r1+i, n] * Bm[b, h*r2+j, n]
__global__ void k_corr(const float* __restrict__ A, const float* __restrict__ Bm,
                       const float* __restrict__ temp, float* __restrict__ out,
                       int heads, int r1, int r2, int Ca, int Cb, int n)
{
  __shared__ float red[256];
  int j = blockIdx.x % r2; int t = blockIdx.x / r2;
  int i = t % r1; t /= r1; int h = t % heads; int b = t / heads;
  const float* ap = A + ((size_t)b * Ca + h * r1 + i) * n;
  const float* bp = Bm + ((size_t)b * Cb + h * r2 + j) * n;
  float s = 0.f;
  for (int k = threadIdx.x; k < n; k += 256) s += ap[k] * bp[k];
  red[threadIdx.x] = s; __syncthreads();
  for (int st = 128; st > 0; st >>= 1) {
    if ((int)threadIdx.x < st) red[threadIdx.x] += red[threadIdx.x + st];
    __syncthreads();
  }
  if (threadIdx.x == 0) out[blockIdx.x] = red[0] * temp[h];
}

__global__ void k_softmax_rows(float* __restrict__ x, int rows, int cols)
{
  int r = blockIdx.x * blockDim.x + threadIdx.x;
  if (r >= rows) return;
  float* p = x + (size_t)r * cols;
  float mx = p[0];
  for (int c = 1; c < cols; ++c) mx = fmaxf(mx, p[c]);
  float s = 0.f;
  for (int c = 0; c < cols; ++c) { float e = expf(p[c] - mx); p[c] = e; s += e; }
  float inv = 1.f / s;
  for (int c = 0; c < cols; ++c) p[c] *= inv;
}

// out[b, h*rout+d, n] = sum_c M[b,h,d,c] * V[b, h*rin+c, n] ; n = 1<<lgN
__global__ void k_attn_apply(const float* __restrict__ M, const float* __restrict__ V,
                             float* __restrict__ out, int Bn, int heads, int rout,
                             int rin, int Cv, int lgN)
{
  int idx = blockIdx.x * blockDim.x + threadIdx.x;
  int n = 1 << lgN;
  int total = (Bn * heads * rout) << lgN;
  if (idx >= total) return;
  int nn = idx & (n - 1); int t = idx >> lgN;
  int d = t % rout; t /= rout; int h = t % heads; int b = t / heads;
  const float* m = M + ((size_t)(b * heads + h) * rout + d) * rin;
  const float* vp = V + (((size_t)b * Cv + h * rin) << lgN) + nn;
  float s = 0.f;
  for (int c = 0; c < rin; ++c) s += m[c] * vp[(size_t)c << lgN];
  out[(((size_t)b * (heads * rout) + h * rout + d) << lgN) + nn] = s;
}

// pixel_unshuffle(2): [B,C,2H,2W] -> [B,4C,H,W]; Hh=Wh=1<<lgS
__global__ void k_unshuffle2(const float* __restrict__ x, float* __restrict__ y,
                             int Bn, int C, int lgS)
{
  int S = 1 << lgS;
  int idx = blockIdx.x * blockDim.x + threadIdx.x;
  int total = Bn * C * 4 * S * S;
  if (idx >= total) return;
  int wo = idx & (S - 1); int t = idx >> lgS;
  int ho = t & (S - 1); t >>= lgS;
  int oc = t % (C * 4); int b = t / (C * 4);
  int c = oc >> 2, ph = (oc >> 1) & 1, pw = oc & 1;
  y[idx] = x[((size_t)((b * C + c) * 2 * S) + 2 * ho + ph) * (2 * S) + 2 * wo + pw];
}

// k x k average pool; Ho=Wo=1<<lgS
__global__ void k_avgpool(const float* __restrict__ x, float* __restrict__ y,
                          int Bn, int C, int lgS, int kpool)
{
  int S = 1 << lgS;
  int idx = blockIdx.x * blockDim.x + threadIdx.x;
  int total = Bn * C * S * S;
  if (idx >= total) return;
  int wo = idx & (S - 1); int t = idx >> lgS;
  int ho = t & (S - 1); t >>= lgS;
  int c = t % C; int b = t / C;
  int Wi = S * kpool;
  const float* xp = x + ((size_t)((b * C + c) * Wi) + ho * kpool) * Wi + wo * kpool;
  float s = 0.f;
  for (int i = 0; i < kpool; ++i)
    for (int j = 0; j < kpool; ++j) s += xp[(size_t)i * Wi + j];
  y[idx] = s / (float)(kpool * kpool);
}

// half-pixel bilinear upsample by s into channel-offset slice of dst (dst 256x256)
__global__ void k_upsample(const float* __restrict__ x, float* __restrict__ y,
                           int Bn, int C, int h, int w, int s, int Ctot, int cOff)
{
  int idx = blockIdx.x * blockDim.x + threadIdx.x;
  int total = Bn * C * 65536;                 // 256*256 output
  if (idx >= total) return;
  int xo = idx & 255; int t = idx >> 8;
  int yo = t & 255; t >>= 8;
  int c = t % C; int b = t / C;
  float sy = (yo + 0.5f) / s - 0.5f;
  float sx = (xo + 0.5f) / s - 0.5f;
  int y0 = (int)floorf(sy), x0 = (int)floorf(sx);
  float fy = sy - y0, fx = sx - x0;
  int y0c = min(max(y0, 0), h - 1), y1c = min(max(y0 + 1, 0), h - 1);
  int x0c = min(max(x0, 0), w - 1), x1c = min(max(x0 + 1, 0), w - 1);
  const float* xp = x + (size_t)(b * C + c) * h * w;
  float v00 = xp[(size_t)y0c * w + x0c], v01 = xp[(size_t)y0c * w + x1c];
  float v10 = xp[(size_t)y1c * w + x0c], v11 = xp[(size_t)y1c * w + x1c];
  float v = v00 * (1 - fy) * (1 - fx) + v01 * (1 - fy) * fx +
            v10 * fy * (1 - fx) + v11 * fy * fx;
  y[((size_t)(b * Ctot + cOff + c) << 16) + (yo << 8) + xo] = v;
}

__global__ void k_copy_ch(const float* __restrict__ src, float* __restrict__ dst,
                          int Bn, int Cs, int lgHW, int Cd, int off)
{
  int HW = 1 << lgHW;
  int idx = blockIdx.x * blockDim.x + threadIdx.x;
  int total = (Bn * Cs) << lgHW;
  if (idx >= total) return;
  int rem = idx & (HW - 1); int t = idx >> lgHW;
  int c = t % Cs; int b = t / Cs;
  dst[(((size_t)b * Cd + off + c) << lgHW) + rem] = src[idx];
}

// encoder input: [x (3ch), mean over channels (1ch)]
__global__ void k_cat_mean(const float* __restrict__ x, float* __restrict__ y,
                           int Bn, int lgHW)
{
  int HW = 1 << lgHW;
  int idx = blockIdx.x * blockDim.x + threadIdx.x;
  if (idx >= (Bn << lgHW)) return;
  int b = idx >> lgHW, rem = idx & (HW - 1);
  const float* xp = x + (((size_t)b * 3) << lgHW) + rem;
  float* yp = y + (((size_t)b * 4) << lgHW) + rem;
  float a0 = xp[0], a1 = xp[(size_t)1 << lgHW], a2 = xp[2 * ((size_t)1 << lgHW)];
  yp[0] = a0; yp[(size_t)1 << lgHW] = a1; yp[2 * ((size_t)1 << lgHW)] = a2;
  yp[3 * ((size_t)1 << lgHW)] = (a0 + a1 + a2) * (1.f / 3.f);
}

__global__ void k_relight(const float* __restrict__ x, const float* __restrict__ m,
                          float* __restrict__ y, int n)
{
  int i = blockIdx.x * blockDim.x + threadIdx.x;
  if (i < n) y[i] = x[i] * m[i] + x[i];
}

__global__ void k_mul_inplace(float* __restrict__ a, const float* __restrict__ b, int n)
{
  int i = blockIdx.x * blockDim.x + threadIdx.x;
  if (i < n) a[i] *= b[i];
}

__global__ void k_zero(float* __restrict__ p, int n)
{
  int i = blockIdx.x * blockDim.x + threadIdx.x;
  if (i < n) p[i] = 0.f;
}

__global__ void k_scale_ch(const float* __restrict__ a, const float* __restrict__ sc,
                           const float* __restrict__ res, float* __restrict__ y,
                           int Bn, int C, int lgHW)
{
  int idx = blockIdx.x * blockDim.x + threadIdx.x;
  int total = (Bn * C) << lgHW;
  if (idx >= total) return;
  int t = idx >> lgHW;
  int c = t % C; int b = t / C;
  y[idx] = a[idx] * sc[b * C + c] + res[idx];
}

__global__ void k_add3(const float* __restrict__ comb, float* __restrict__ s, int sz)
{
  int i = blockIdx.x * blockDim.x + threadIdx.x;
  if (i < sz) s[i] = comb[i] + comb[i + (size_t)sz] + comb[i + 2 * (size_t)sz];
}

__global__ void k_softmax_branch(float* __restrict__ z, int Bn)
{
  int idx = blockIdx.x * blockDim.x + threadIdx.x;
  if (idx >= Bn * 32) return;
  int c = idx % 32, b = idx / 32;
  float v0 = z[b * 96 + c], v1 = z[b * 96 + 32 + c], v2 = z[b * 96 + 64 + c];
  float mx = fmaxf(v0, fmaxf(v1, v2));
  float e0 = expf(v0 - mx), e1 = expf(v1 - mx), e2 = expf(v2 - mx);
  float inv = 1.f / (e0 + e1 + e2);
  z[b * 96 + c] = e0 * inv; z[b * 96 + 32 + c] = e1 * inv; z[b * 96 + 64 + c] = e2 * inv;
}

__global__ void k_mix(const float* __restrict__ comb, const float* __restrict__ wts,
                      float* __restrict__ out, int Bn, int C, int lgHW)
{
  int idx = blockIdx.x * blockDim.x + threadIdx.x;
  int total = (Bn * C) << lgHW;
  if (idx >= total) return;
  int t = idx >> lgHW;
  int c = t % C; int b = t / C;
  size_t sz = (size_t)total;
  out[idx] = comb[idx] * wts[b * 96 + c] + comb[idx + sz] * wts[b * 96 + 32 + c] +
             comb[idx + 2 * sz] * wts[b * 96 + 64 + c];
}

// psi += gate * wg * exp(-0.5*((xi-mu)/(sigma*tau+eps))^2); p channels = c*5+j
__global__ void k_gauss(float* __restrict__ psi, const float* __restrict__ p,
                        const float* __restrict__ xi, int Bn, int lgHW)
{
  int idx = blockIdx.x * blockDim.x + threadIdx.x;
  int total = (Bn * 3) << lgHW;
  if (idx >= total) return;
  int HW = 1 << lgHW;
  int rem = idx & (HW - 1); int t = idx >> lgHW;
  int c = t % 3; int b = t / 3;
  const float* pp = p + (((size_t)b * 15 + c * 5) << lgHW) + rem;
  float wg   = tanhf(pp[0]) * 1.5f;
  float mu   = tanhf(pp[(size_t)1 << lgHW]) * 4.0f;
  float sg   = 1.f / (1.f + expf(-pp[2 * ((size_t)1 << lgHW)])) * 4.0f + 0.05f;
  float gate = 1.f / (1.f + expf(-pp[3 * ((size_t)1 << lgHW)]));
  float tau  = 1.f / (1.f + expf(-pp[4 * ((size_t)1 << lgHW)])) * 2.0f + 0.2f;
  float d = (xi[idx] - mu) / (sg * tau + 1e-6f);
  psi[idx] += gate * wg * expf(-0.5f * d * d);
}

// ---------------------------------------------------------------------------
// Host-side orchestration
// ---------------------------------------------------------------------------
struct Conv { const float* w; const float* b; };
struct Ln   { const float* g; const float* b; };
struct Ffn  { Conv dw, pw1, pw2; };                 // flatten order: dw, pw1, pw2
struct Dkan { Conv dw, pin, pout; };                // flatten order: dw, pin, pout
struct Attn {
  Conv a1, a2, k, out, q;                           // a1,a2,k,out,q
  const float* temp_a; const float* temp_v;         // temp_a, temp_v
  Conv v;                                           // v
};
struct TB { Attn attn; Ffn ffn; Ln norm1, norm2; }; // attn, ffn, norm1, norm2

struct Arena {
  char* base; size_t off;
  float* alloc(size_t nfloats) {
    float* p = (float*)(base + off);
    off += ((nfloats * sizeof(float)) + 255) & ~(size_t)255;
    return p;
  }
};

static inline unsigned gsz(long total) { return (unsigned)((total + 255) / 256); }
static inline int cdiv(long a, long b) { return (int)((a + b - 1) / b); }
static inline int lg2i(long v) { int l = 0; while ((1L << l) < v) ++l; return l; }

template <int KS, bool KF, bool MF>
static void pw_launch(hipStream_t s, dim3 grid, const float* X, Conv c,
                      const float* Rin, float* Y, int Cin, int Cout, int lgHW,
                      int Bn, int act)
{
  if (Rin)
    k_pw_wmma<KS, KF, MF, true><<<grid, 128, 0, s>>>(X, c.w, c.b, Rin, Y, Cin, Cout, lgHW, Bn, act);
  else
    k_pw_wmma<KS, KF, MF, false><<<grid, 128, 0, s>>>(X, c.w, c.b, Rin, Y, Cin, Cout, lgHW, Bn, act);
}

static void pw_wmma(hipStream_t s, const float* X, Conv c, const float* Rin, float* Y,
                    int Cin, int Cout, int HW, int Bn, int act)
{
  int lgHW = lg2i(HW);
  long Ntot = (long)Bn * HW;                        // multiple of 64
  dim3 grid((unsigned)cdiv(Ntot >> 6, 4), (unsigned)cdiv(Cout, 16));
  bool kf = (Cin % 32) == 0;
  bool mf = (Cout % 16) == 0;
  if (Cin <= 32) {
    if (kf && mf)       pw_launch<1, true,  true >(s, grid, X, c, Rin, Y, Cin, Cout, lgHW, Bn, act);
    else if (kf)        pw_launch<1, true,  false>(s, grid, X, c, Rin, Y, Cin, Cout, lgHW, Bn, act);
    else if (mf)        pw_launch<1, false, true >(s, grid, X, c, Rin, Y, Cin, Cout, lgHW, Bn, act);
    else                pw_launch<1, false, false>(s, grid, X, c, Rin, Y, Cin, Cout, lgHW, Bn, act);
  } else {
    if (kf && mf)       pw_launch<2, true,  true >(s, grid, X, c, Rin, Y, Cin, Cout, lgHW, Bn, act);
    else if (kf)        pw_launch<2, true,  false>(s, grid, X, c, Rin, Y, Cin, Cout, lgHW, Bn, act);
    else if (mf)        pw_launch<2, false, true >(s, grid, X, c, Rin, Y, Cin, Cout, lgHW, Bn, act);
    else                pw_launch<2, false, false>(s, grid, X, c, Rin, Y, Cin, Cout, lgHW, Bn, act);
  }
}

static void conv2d(hipStream_t s, const float* X, Conv c, float* Y, int Bn, int Ci,
                   int H, int W, int Co, int K, int stride, int pad, int dil,
                   int groups, int act)
{
  int Ho = (H + 2 * pad - dil * (K - 1) - 1) / stride + 1;
  int Wo = (W + 2 * pad - dil * (K - 1) - 1) / stride + 1;
  long total = (long)Bn * Co * Ho * Wo;
  k_conv2d<<<gsz(total), 256, 0, s>>>(X, c.w, c.b, Y, Bn, Ci, H, W, Co, K,
                                      stride, pad, dil, groups, act, Ho, Wo);
}

static void run_tblock(hipStream_t s, Arena& ar, const float* x, const float* illu,
                       const TB& p, int Bn, int C, int H, int heads, float* yout)
{
  size_t mk = ar.off;
  int nf = H * H; int lgnf = lg2i(nf);
  int Hs = H / 2; int ns = Hs * Hs;
  int ch = C / heads, ch2 = ch / 2;

  float* ln1 = ar.alloc((size_t)Bn * C * nf);
  k_layernorm<<<gsz((long)Bn * nf), 256, 0, s>>>(x, p.norm1.g, p.norm1.b, ln1, Bn, C, lgnf);

  float* q_  = ar.alloc((size_t)Bn * C * ns);
  float* kx  = ar.alloc((size_t)Bn * C * ns);
  float* a1t = ar.alloc((size_t)Bn * C * ns);
  float* aa  = ar.alloc((size_t)Bn * (C / 2) * ns);
  float* v_  = ar.alloc((size_t)Bn * C * nf);
  conv2d(s, ln1, p.attn.q,  q_,  Bn, C, H, H, C, 3, 2, 1, 1, C, ACT_NONE);
  conv2d(s, ln1, p.attn.k,  kx,  Bn, C, H, H, C, 3, 2, 1, 1, 1, ACT_NONE);
  conv2d(s, ln1, p.attn.a1, a1t, Bn, C, H, H, C, 3, 2, 1, 1, C, ACT_NONE);
  pw_wmma(s, a1t, p.attn.a2, nullptr, aa, C, C / 2, ns, Bn, ACT_NONE);
  pw_wmma(s, ln1, p.attn.v,  nullptr, v_, C, C,     nf, Bn, ACT_NONE);
  k_mul_inplace<<<gsz((long)Bn * C * nf), 256, 0, s>>>(v_, illu, Bn * C * nf);

  k_l2norm_rows<<<Bn * C,       256, 0, s>>>(q_, ns);
  k_l2norm_rows<<<Bn * C,       256, 0, s>>>(kx, ns);
  k_l2norm_rows<<<Bn * (C / 2), 256, 0, s>>>(aa, ns);

  float* attn_a = ar.alloc((size_t)Bn * heads * ch * ch2);
  float* attn_k = ar.alloc((size_t)Bn * heads * ch2 * ch);
  k_corr<<<Bn * heads * ch * ch2, 256, 0, s>>>(q_, aa, p.attn.temp_a, attn_a,
                                               heads, ch, ch2, C, C / 2, ns);
  k_corr<<<Bn * heads * ch2 * ch, 256, 0, s>>>(aa, kx, p.attn.temp_v, attn_k,
                                               heads, ch2, ch, C / 2, C, ns);
  k_softmax_rows<<<gsz(Bn * heads * ch),  64, 0, s>>>(attn_a, Bn * heads * ch, ch2);
  k_softmax_rows<<<gsz(Bn * heads * ch2), 64, 0, s>>>(attn_k, Bn * heads * ch2, ch);

  float* tmp = ar.alloc((size_t)Bn * (C / 2) * nf);
  float* att = ar.alloc((size_t)Bn * C * nf);
  k_attn_apply<<<gsz((long)Bn * heads * ch2 * nf), 256, 0, s>>>(attn_k, v_, tmp, Bn,
                                                                heads, ch2, ch, C, lgnf);
  k_attn_apply<<<gsz((long)Bn * heads * ch * nf), 256, 0, s>>>(attn_a, tmp, att, Bn,
                                                               heads, ch, ch2, C / 2, lgnf);
  float* y1 = ar.alloc((size_t)Bn * C * nf);
  pw_wmma(s, att, p.attn.out, x /*residual*/, y1, C, C, nf, Bn, ACT_NONE);

  float* ln2 = ar.alloc((size_t)Bn * C * nf);
  k_layernorm<<<gsz((long)Bn * nf), 256, 0, s>>>(y1, p.norm2.g, p.norm2.b, ln2, Bn, C, lgnf);
  float* f1 = ar.alloc((size_t)Bn * C * nf);
  float* f2 = ar.alloc((size_t)Bn * C * nf);
  pw_wmma(s, ln2, p.ffn.pw1, nullptr, f1, C, C, nf, Bn, ACT_NONE);
  conv2d(s, f1, p.ffn.dw, f2, Bn, C, H, H, C, 3, 1, 1, 1, C, ACT_SILU);
  pw_wmma(s, f2, p.ffn.pw2, y1 /*residual*/, yout, C, C, nf, Bn, ACT_NONE);
  ar.off = mk;
}

extern "C" void kernel_launch(void* const* d_in, const int* in_sizes, int n_in,
                              void* d_out, int out_size, void* d_ws, size_t ws_size,
                              hipStream_t stream)
{
  (void)in_sizes; (void)n_in; (void)out_size; (void)ws_size;

  // -------- parameter unpacking (jax tree order: dict keys sorted, tuples in order)
  int ii = 0;
  auto F = [&](int k) -> const float* { return (const float*)d_in[k]; };
  const float* Xin = F(ii++);                              // 'x'
  auto CONV = [&]() -> Conv { Conv c{F(ii), F(ii + 1)}; ii += 2; return c; };
  auto LNP  = [&]() -> Ln   { Ln l{F(ii), F(ii + 1)}; ii += 2; return l; };
  auto FFNP = [&]() -> Ffn  { Ffn f; f.dw = CONV(); f.pw1 = CONV(); f.pw2 = CONV(); return f; };
  auto DKANP= [&]() -> Dkan { Dkan d; d.dw = CONV(); d.pin = CONV(); d.pout = CONV(); return d; };
  auto ATTNP= [&]() -> Attn {
    Attn a; a.a1 = CONV(); a.a2 = CONV(); a.k = CONV(); a.out = CONV(); a.q = CONV();
    a.temp_a = F(ii++); a.temp_v = F(ii++); a.v = CONV(); return a;
  };
  auto TBP  = [&]() -> TB {
    TB t; t.attn = ATTNP(); t.ffn = FFNP(); t.norm1 = LNP(); t.norm2 = LNP(); return t;
  };
  // params: chi, enc, heads, x_norm, xi
  Dkan b1 = DKANP(), b2 = DKANP(), b4 = DKANP();           // chi: b1,b2,b4
  Conv fc1 = CONV(), fc2 = CONV(), post = CONV(), pre = CONV();
  Ffn co_ffn = FFNP(); Ln co_ln = LNP();                   // enc: co_ffn, co_ln
  Conv est_c1 = CONV(), est_c2 = CONV(), est_dw = CONV();
  Conv illu_d1 = CONV(), illu_d2 = CONV();
  TB trans1 = TBP(), trans2 = TBP();
  Ffn heads_p[8];
  for (int g = 0; g < 8; ++g) heads_p[g] = FFNP();         // heads list
  Ln x_norm = LNP();                                       // x_norm
  Conv xc1 = CONV(), xc2 = CONV(), cg1 = CONV(), cg2 = CONV(); // xi: c1,c2,cg1,cg2
  Ln xln = LNP(), xln_out = LNP(); Conv xlocal = CONV();       // ln, ln_out, local

  const int Bn = 2, Himg = 256;
  const int HW = 65536, lgHW = 16;
  Arena ar{(char*)d_ws, 0};

  // persistent buffers
  float* xi_out = ar.alloc((size_t)Bn * 3 * HW);
  float* psi    = ar.alloc((size_t)Bn * 3 * HW);
  float* feat   = ar.alloc((size_t)Bn * 32 * HW);
  float* x0     = ar.alloc((size_t)Bn * 3 * HW);
  size_t mark0 = ar.off;

  // ---------------- xi_net ----------------
  {
    float* t16a = ar.alloc((size_t)Bn * 16 * HW);  // c1 out (residual)
    float* t16b = ar.alloc((size_t)Bn * 16 * HW);  // xn / combined
    float* t16c = ar.alloc((size_t)Bn * 16 * HW);  // dw local out
    float* v16a = ar.alloc(Bn * 16);
    float* v16b = ar.alloc(Bn * 16);
    float* v16c = ar.alloc(Bn * 16);
    float* t3a  = ar.alloc((size_t)Bn * 3 * HW);
    conv2d(stream, Xin, xc1, t16a, Bn, 3, Himg, Himg, 16, 3, 1, 1, 1, 1, ACT_NONE);
    k_layernorm<<<gsz((long)Bn * HW), 256, 0, stream>>>(t16a, xln.g, xln.b, t16b, Bn, 16, lgHW);
    k_spatial_mean<<<Bn * 16, 256, 0, stream>>>(t16b, v16a, HW);
    k_vec_conv<<<1, 64, 0, stream>>>(v16a, cg1.w, cg1.b, v16b, Bn, 16, 16, ACT_SILU);
    k_vec_conv<<<1, 64, 0, stream>>>(v16b, cg2.w, cg2.b, v16c, Bn, 16, 16, ACT_SIGM);
    conv2d(stream, t16b, xlocal, t16c, Bn, 16, Himg, Himg, 16, 3, 1, 1, 1, 16, ACT_NONE);
    k_scale_ch<<<gsz((long)Bn * 16 * HW), 256, 0, stream>>>(t16c, v16c, t16a, t16b, Bn, 16, lgHW);
    pw_wmma(stream, t16b, xc2, nullptr, t3a, 16, 3, HW, Bn, ACT_NONE);
    k_layernorm<<<gsz((long)Bn * HW), 256, 0, stream>>>(t3a, xln_out.g, xln_out.b, xi_out, Bn, 3, lgHW);
    ar.off = mark0;
  }

  // ---------------- encoder ----------------
  {
    size_t mkE = ar.off;
    float* cat4 = ar.alloc((size_t)Bn * 4 * HW);
    k_cat_mean<<<gsz((long)Bn * HW), 256, 0, stream>>>(Xin, cat4, Bn, lgHW);
    float* fea0 = ar.alloc((size_t)Bn * 12 * HW);
    pw_wmma(stream, cat4, est_c1, nullptr, fea0, 4, 12, HW, Bn, ACT_NONE);
    float* fea = ar.alloc((size_t)Bn * 12 * HW);
    conv2d(stream, fea0, est_dw, fea, Bn, 12, Himg, Himg, 12, 5, 1, 2, 1, 4, ACT_NONE);
    float* imap = ar.alloc((size_t)Bn * 3 * HW);
    pw_wmma(stream, fea, est_c2, nullptr, imap, 12, 3, HW, Bn, ACT_NONE);
    k_relight<<<gsz((long)Bn * 3 * HW), 256, 0, stream>>>(Xin, imap, x0, Bn * 3 * HW);

    const int H1 = 128; const int HW1 = H1 * H1;
    float* x1in  = ar.alloc((size_t)Bn * 12 * HW1);
    k_unshuffle2<<<gsz((long)Bn * 12 * HW1), 256, 0, stream>>>(x0, x1in, Bn, 3, 7);
    float* fea2  = ar.alloc((size_t)Bn * 12 * HW1);
    k_avgpool<<<gsz((long)Bn * 12 * HW1), 256, 0, stream>>>(fea, fea2, Bn, 12, 7, 2);
    float* illu1 = ar.alloc((size_t)Bn * 12 * HW1);
    pw_wmma(stream, fea2, illu_d1, nullptr, illu1, 12, 12, HW1, Bn, ACT_NONE);
    float* x1out = ar.alloc((size_t)Bn * 12 * HW1);
    run_tblock(stream, ar, x1in, illu1, trans1, Bn, 12, H1, 3, x1out);

    const int H2 = 64; const int HW2 = H2 * H2;
    float* x2in  = ar.alloc((size_t)Bn * 48 * HW2);
    k_unshuffle2<<<gsz((long)Bn * 48 * HW2), 256, 0, stream>>>(x1out, x2in, Bn, 12, 6);
    float* fea4  = ar.alloc((size_t)Bn * 12 * HW2);
    k_avgpool<<<gsz((long)Bn * 12 * HW2), 256, 0, stream>>>(fea, fea4, Bn, 12, 6, 4);
    float* illu2 = ar.alloc((size_t)Bn * 48 * HW2);
    pw_wmma(stream, fea4, illu_d2, nullptr, illu2, 12, 48, HW2, Bn, ACT_NONE);
    float* x2out = ar.alloc((size_t)Bn * 48 * HW2);
    run_tblock(stream, ar, x2in, illu2, trans2, Bn, 48, H2, 3, x2out);

    float* cat63 = ar.alloc((size_t)Bn * 63 * HW);
    k_copy_ch<<<gsz((long)Bn * 3 * HW), 256, 0, stream>>>(x0, cat63, Bn, 3, lgHW, 63, 0);
    k_upsample<<<gsz((long)Bn * 12 * HW), 256, 0, stream>>>(x1out, cat63, Bn, 12, H1, H1, 2, 63, 3);
    k_upsample<<<gsz((long)Bn * 48 * HW), 256, 0, stream>>>(x2out, cat63, Bn, 48, H2, H2, 4, 63, 15);
    float* catln = ar.alloc((size_t)Bn * 63 * HW);
    k_layernorm<<<gsz((long)Bn * HW), 256, 0, stream>>>(cat63, co_ln.g, co_ln.b, catln, Bn, 63, lgHW);
    float* cf1 = ar.alloc((size_t)Bn * 32 * HW);
    float* cf2 = ar.alloc((size_t)Bn * 32 * HW);
    pw_wmma(stream, catln, co_ffn.pw1, nullptr, cf1, 63, 32, HW, Bn, ACT_NONE);
    conv2d(stream, cf1, co_ffn.dw, cf2, Bn, 32, Himg, Himg, 32, 3, 1, 1, 1, 32, ACT_SILU);
    pw_wmma(stream, cf2, co_ffn.pw2, nullptr, feat, 32, 32, HW, Bn, ACT_NONE);
    ar.off = mkE;
  }

  // ---------------- 8 gaussian heads -> psi ----------------
  {
    k_zero<<<gsz((long)Bn * 3 * HW), 256, 0, stream>>>(psi, Bn * 3 * HW);
    size_t mkH = ar.off;
    float* h1 = ar.alloc((size_t)Bn * 32 * HW);
    float* h2 = ar.alloc((size_t)Bn * 32 * HW);
    float* hp = ar.alloc((size_t)Bn * 15 * HW);
    for (int g = 0; g < 8; ++g) {
      pw_wmma(stream, feat, heads_p[g].pw1, nullptr, h1, 32, 32, HW, Bn, ACT_NONE);
      conv2d(stream, h1, heads_p[g].dw, h2, Bn, 32, Himg, Himg, 32, 3, 1, 1, 1, 32, ACT_SILU);
      pw_wmma(stream, h2, heads_p[g].pw2, nullptr, hp, 32, 15, HW, Bn, ACT_NONE);
      k_gauss<<<gsz((long)Bn * 3 * HW), 256, 0, stream>>>(psi, hp, xi_out, Bn, lgHW);
    }
    ar.off = mkH;
  }

  // ---------------- chi_net ----------------
  {
    float* xn3 = ar.alloc((size_t)Bn * 3 * HW);
    k_layernorm<<<gsz((long)Bn * HW), 256, 0, stream>>>(Xin, x_norm.g, x_norm.b, xn3, Bn, 3, lgHW);
    float* cat9 = ar.alloc((size_t)Bn * 9 * HW);
    k_copy_ch<<<gsz((long)Bn * 3 * HW), 256, 0, stream>>>(Xin, cat9, Bn, 3, lgHW, 9, 0);
    k_copy_ch<<<gsz((long)Bn * 3 * HW), 256, 0, stream>>>(xn3, cat9, Bn, 3, lgHW, 9, 3);
    k_copy_ch<<<gsz((long)Bn * 3 * HW), 256, 0, stream>>>(psi, cat9, Bn, 3, lgHW, 9, 6);
    float* c0 = ar.alloc((size_t)Bn * 32 * HW);
    pw_wmma(stream, cat9, pre, nullptr, c0, 9, 32, HW, Bn, ACT_NONE);
    float* comb = ar.alloc((size_t)3 * Bn * 32 * HW);
    float* dt1  = ar.alloc((size_t)Bn * 32 * HW);
    float* dt2  = ar.alloc((size_t)Bn * 32 * HW);
    const Dkan* dk[3] = {&b1, &b2, &b4};
    const int dil[3] = {1, 2, 4};
    for (int br = 0; br < 3; ++br) {
      pw_wmma(stream, c0, dk[br]->pin, nullptr, dt1, 32, 32, HW, Bn, ACT_NONE);
      conv2d(stream, dt1, dk[br]->dw, dt2, Bn, 32, Himg, Himg, 32, 3, 1, dil[br], dil[br], 32, ACT_SILU);
      pw_wmma(stream, dt2, dk[br]->pout, c0 /*residual*/, comb + (size_t)br * Bn * 32 * HW,
              32, 32, HW, Bn, ACT_NONE);
    }
    float* ssum = dt1;  // reuse
    k_add3<<<gsz((long)Bn * 32 * HW), 256, 0, stream>>>(comb, ssum, Bn * 32 * HW);
    float* sm = ar.alloc(Bn * 32);
    float* z1 = ar.alloc(Bn * 8);
    float* z2 = ar.alloc(Bn * 96);
    k_spatial_mean<<<Bn * 32, 256, 0, stream>>>(ssum, sm, HW);
    k_vec_conv<<<1, 64, 0, stream>>>(sm, fc1.w, fc1.b, z1, Bn, 32, 8, ACT_SILU);
    k_vec_conv<<<1, 256, 0, stream>>>(z1, fc2.w, fc2.b, z2, Bn, 8, 96, ACT_NONE);
    k_softmax_branch<<<1, 64, 0, stream>>>(z2, Bn);
    float* mix = dt2;  // reuse
    k_mix<<<gsz((long)Bn * 32 * HW), 256, 0, stream>>>(comb, z2, mix, Bn, 32, lgHW);
    pw_wmma(stream, mix, post, nullptr, (float*)d_out, 32, 3, HW, Bn, ACT_NONE);
  }
}
